// Axial_Layer_cross_54966991454518
// MI455X (gfx1250) — matrix operand order repaired and checked by reading, not compiled, not run on hardware
//
#include <hip/hip_runtime.h>
#include <hip/hip_bf16.h>

// Axial attention (cross) for MI455X / gfx1250, wave32 + WMMA bf16 + TDM.
// Pipeline:
//   k_prep    : gather rel_enc -> rel(128,64,64), zero stat accumulators
//   k_pack    : weights -> bf16 wcat(1024,512); x/y -> bf16 xb/yb[n][h][c]
//   k_proj    : kqv_raw(N,1024,64) bf16 via v_wmma_f32_16x16x32_bf16;
//               tiles staged in LDS by TENSOR_LOAD_TO_LDS (TDM) when available
//   k_chstats : fold BN into per-channel scale/bias (1024)
//   k_lstats  : qk (WMMA) + qr/kr (VALU) sum/sumsq per 24 logit channels
//   k_lscale  : a[j,h] = lg_gamma / sqrt(var+eps)   (means/betas cancel in softmax)
//   k_attn    : recompute scores, softmax, attn = W*v^T (WMMA) + attn_enc (VALU)

typedef __bf16 bf16_t;
typedef bf16_t v16bf __attribute__((ext_vector_type(16)));
typedef bf16_t v8bf  __attribute__((ext_vector_type(8)));
typedef float  v8f   __attribute__((ext_vector_type(8)));
typedef unsigned int v4u __attribute__((ext_vector_type(4)));
typedef int v4i __attribute__((ext_vector_type(4)));
typedef int v8i __attribute__((ext_vector_type(8)));

#if defined(__has_builtin)
#if __has_builtin(__builtin_amdgcn_tensor_load_to_lds) && \
    __has_builtin(__builtin_amdgcn_s_wait_tensorcnt)
#define USE_TDM 1
#endif
#endif
#ifndef USE_TDM
#define USE_TDM 0
#endif

union Frag {
  unsigned short u[16];
  struct { v8bf lo, hi; } p;
  v16bf v;
};
union U4 { uint4 q; unsigned short u[8]; };

__device__ __forceinline__ unsigned short f2bf(float f) {
  union { float f; unsigned u; } c; c.f = f;
  unsigned uu = c.u;
  unsigned r = uu + 0x7FFFu + ((uu >> 16) & 1u);   // round-to-nearest-even
  return (unsigned short)(r >> 16);
}
__device__ __forceinline__ float bf2f(unsigned short h) {
  union { unsigned u; float f; } c; c.u = ((unsigned)h) << 16;
  return c.f;
}
__device__ __forceinline__ v8f wmma_bf16(v16bf a, v16bf b, v8f c) {
  return __builtin_amdgcn_wmma_f32_16x16x32_bf16(false, a, false, b, (short)0, c,
                                                 false, false);
}
__device__ __forceinline__ v16bf ld16(const unsigned short* ptr) {
  return *(const v16bf*)ptr;
}
__device__ __forceinline__ v16bf ld8x2(const unsigned short* plo,
                                       const unsigned short* phi) {
  Frag f;
  f.p.lo = *(const v8bf*)plo;
  f.p.hi = *(const v8bf*)phi;
  return f.v;
}

#if USE_TDM
__device__ __forceinline__ unsigned lds_offset(const void* p) {
  // ptrtoint of an AS(3) pointer == byte offset within the workgroup LDS
  return (unsigned)(unsigned long long)(__attribute__((address_space(3))) const void*)p;
}
// 2-D bf16 tile load: d0 contiguous elements per row, d1 rows, row stride in elems.
// D# fields per CDNA5 ISA ch.8 (count=1, data_size=2B, type=2 "image").
__device__ __forceinline__ void tdm_load_2d(unsigned lds, unsigned long long gaddr,
                                            unsigned d0, unsigned d1,
                                            unsigned stride) {
  v4u g0;
  g0.x = 1u;                                                   // count=1, user mode
  g0.y = lds;                                                  // lds_addr (bytes)
  g0.z = (unsigned)gaddr;                                      // global_addr[31:0]
  g0.w = ((unsigned)(gaddr >> 32) & 0x01FFFFFFu) | (2u << 30); // addr[56:32] | type=2
  v8i g1;
  g1[0] = (int)(1u << 16);                                     // data_size = 1 (2B)
  g1[1] = (int)((d0 & 0xFFFFu) << 16);                         // tensor_dim0[15:0]
  g1[2] = (int)(((d0 >> 16) & 0xFFFFu) | ((d1 & 0xFFFFu) << 16)); // dim0 hi | dim1 lo
  g1[3] = (int)(((d1 >> 16) & 0xFFFFu) | ((d0 & 0xFFFFu) << 16)); // dim1 hi | tile_dim0
  g1[4] = (int)(d1 & 0xFFFFu);                                 // tile_dim1, tile_dim2=0
  g1[5] = (int)stride;                                         // tensor_dim0_stride lo
  g1[6] = 0;                                                   // stride hi | dim1_stride lo
  g1[7] = 0;
  v4i g2 = (v4i){};
  v4i g3 = (v4i){};
#if defined(__clang_major__) && (__clang_major__ >= 23)
  v8i g4 = (v8i){};
  __builtin_amdgcn_tensor_load_to_lds(g0, g1, g2, g3, g4, 0);
#else
  __builtin_amdgcn_tensor_load_to_lds(g0, g1, g2, g3, 0);
#endif
}
#endif  // USE_TDM

// ---------------------------------------------------------------- k_prep
__global__ void __launch_bounds__(256) k_prep(const float* __restrict__ rel_enc,
                                              float* __restrict__ rel,
                                              float* __restrict__ stats) {
  int i = blockIdx.x * 256 + threadIdx.x;
  if (i < 2096) stats[i] = 0.f;                       // gsum/gsq/lsum/lsq
  if (i < 128 * 64 * 64) {
    int d = i >> 12, r = i & 4095, xx = r >> 6, yy = r & 63;
    rel[i] = rel_enc[d * 127 + (yy - xx + 63)];
  }
}

// ---------------------------------------------------------------- k_pack
// wb: bf16 wcat(1024,512); xb/yb: bf16 [n][h][c] (TDM-friendly: c contiguous)
__global__ void __launch_bounds__(256) k_pack(
    const float* __restrict__ x, const float* __restrict__ y,
    const float* __restrict__ k_w, const float* __restrict__ q_w,
    const float* __restrict__ v_w, unsigned short* __restrict__ wb,
    unsigned short* __restrict__ xb, unsigned short* __restrict__ yb) {
  long long i = (long long)blockIdx.x * 256 + threadIdx.x;
  if (i < 524288) {                                   // weights 1024x512
    int c = (int)(i >> 9), kl = (int)(i & 511);
    float v;
    if (c < 256)      v = k_w[c * 512 + kl];
    else if (c < 512) v = q_w[(c - 256) * 512 + kl];
    else              v = v_w[(c - 512) * 512 + kl];
    wb[i] = f2bf(v);
  } else {
    long long j = i - 524288;                         // 2 x 16,777,216 activations
    const float* src = x;
    unsigned short* dst = xb;
    if (j >= 16777216) { j -= 16777216; src = y; dst = yb; }
    int n = (int)(j >> 15), r = (int)(j & 32767);
    int h = r >> 9, c = r & 511;
    int b = n >> 8, w = n & 255;
    dst[j] = f2bf(src[((b * 512 + c) * 64 + h) * 256 + w]);
  }
}

// ---------------------------------------------------------------- k_proj
// grid (512, 4): block = (n, channel-group of 256 cat-channels)
__global__ void __launch_bounds__(256) k_proj(
    const unsigned short* __restrict__ wb, const unsigned short* __restrict__ xb,
    const unsigned short* __restrict__ yb, unsigned short* __restrict__ kqv,
    float* __restrict__ gsum, float* __restrict__ gsq) {
  const int n = blockIdx.x, g = blockIdx.y;
  const int t = threadIdx.x, lane = t & 31, wave = t >> 5;
  const int mcol = lane & 15, half = lane >> 4;
  const int kbA = (lane < 16) ? 0 : 16;
  const int kbB = (lane < 16) ? 0 : 16;

  const unsigned short* Xrow = ((g == 1) ? yb : xb) + n * 32768;  // [h][c]
  const int catBase = g << 8;
  const unsigned short* Wrow = wb + catBase * 512;

  __shared__ __attribute__((aligned(32))) unsigned short lA[256 * 32];  // [ch][k]
  __shared__ __attribute__((aligned(32))) unsigned short lBt[64 * 32];  // [h][k]
  __shared__ float csum[256];
  __shared__ float csq[256];
  csum[t] = 0.f; csq[t] = 0.f;

#if USE_TDM
  const unsigned ldsA = lds_offset(lA);
  const unsigned ldsB = lds_offset(lBt);
#endif

  v8f acc[2][4];
#pragma unroll
  for (int rt = 0; rt < 2; ++rt)
#pragma unroll
    for (int ht = 0; ht < 4; ++ht) acc[rt][ht] = (v8f){};

  for (int kk = 0; kk < 512; kk += 32) {
    __syncthreads();
#if USE_TDM
    if (t < 32) {                                     // wave 0 drives the DMA
      tdm_load_2d(ldsA, (unsigned long long)(Wrow + kk), 32u, 256u, 512u);
      tdm_load_2d(ldsB, (unsigned long long)(Xrow + kk), 32u, 64u, 512u);
      __builtin_amdgcn_s_wait_tensorcnt(0);
    }
#else
#pragma unroll
    for (int i = 0; i < 4; ++i) {                     // A: 1024 chunks of 8 halves
      int m2 = i * 256 + t;
      int ch = m2 >> 2, koff = (m2 & 3) * 8;
      *(uint4*)&lA[ch * 32 + koff] = *(const uint4*)&Wrow[ch * 512 + kk + koff];
    }
    {
      int h = t >> 2, koff = (t & 3) * 8;             // B: 256 chunks of 8 halves
      *(uint4*)&lBt[h * 32 + koff] = *(const uint4*)&Xrow[h * 512 + kk + koff];
    }
#endif
    __syncthreads();

    v16bf av[2];
#pragma unroll
    for (int rt = 0; rt < 2; ++rt) {
      int row = wave * 32 + rt * 16 + mcol;
      av[rt] = ld8x2(&lA[row * 32 + kbA], &lA[row * 32 + kbA + 8]);
    }
#pragma unroll
    for (int ht = 0; ht < 4; ++ht) {
      v16bf bv = ld16(&lBt[(ht * 16 + mcol) * 32 + kbB]);
      acc[0][ht] = wmma_bf16(av[0], bv, acc[0][ht]);
      acc[1][ht] = wmma_bf16(av[1], bv, acc[1][ht]);
    }
  }

#pragma unroll
  for (int rt = 0; rt < 2; ++rt)
#pragma unroll
    for (int r = 0; r < 8; ++r) {
      int c = wave * 32 + rt * 16 + half * 8 + r;     // local channel 0..255
      float s = 0.f, sq = 0.f;
#pragma unroll
      for (int ht = 0; ht < 4; ++ht) {
        float vv = acc[rt][ht][r];
        int h = ht * 16 + mcol;
        kqv[(n * 1024 + catBase + c) * 64 + h] = f2bf(vv);
        s += vv; sq += vv * vv;
      }
      atomicAdd(&csum[c], s);
      atomicAdd(&csq[c], sq);
    }
  __syncthreads();
  atomicAdd(&gsum[catBase + t], csum[t]);
  atomicAdd(&gsq[catBase + t], csq[t]);
}

// ---------------------------------------------------------------- k_chstats
__global__ void k_chstats(const float* __restrict__ gsum, const float* __restrict__ gsq,
                          const float* __restrict__ kg, const float* __restrict__ kbet,
                          const float* __restrict__ qg, const float* __restrict__ qb,
                          const float* __restrict__ vg, const float* __restrict__ vb,
                          float* __restrict__ scale, float* __restrict__ bias) {
  int c = blockIdx.x * 256 + threadIdx.x;
  if (c >= 1024) return;
  const float inv = 1.f / 32768.f;                    // N*H
  float m = gsum[c] * inv;
  float var = gsq[c] * inv - m * m;
  float g, be;
  if (c < 256)      { g = kg[c];       be = kbet[c]; }
  else if (c < 512) { g = qg[c - 256]; be = qb[c - 256]; }
  else              { g = vg[c - 512]; be = vb[c - 512]; }  // v uses kq_bn params
  float s = g * rsqrtf(var + 1e-5f);
  scale[c] = s;
  bias[c]  = be - m * s;
}

// ---------------------------------------------------------------- k_lstats
// grid (512, 8): block = (n, head). Accumulate sum/sumsq of qk/qr/kr logits.
__global__ void __launch_bounds__(256) k_lstats(
    const unsigned short* __restrict__ kqv, const float* __restrict__ scale,
    const float* __restrict__ bias, const float* __restrict__ rel,
    float* __restrict__ lsum, float* __restrict__ lsq) {
  const int n = blockIdx.x, head = blockIdx.y;
  const int t = threadIdx.x, lane = t & 31, wave = t >> 5;
  const int mcol = lane & 15;
  const int kbA = (lane < 16) ? 0 : 16;
  const int kbB = (lane < 16) ? 0 : 16;

  __shared__ float qs[2048];                                         // q[d][h]
  __shared__ float ks[2048];                                         // k[d][h]
  __shared__ __attribute__((aligned(32))) unsigned short qsT[2048];  // [x][d] bf16
  __shared__ __attribute__((aligned(32))) unsigned short ksT[2048];  // [y][d] bf16
  __shared__ float red[6];
  if (t < 6) red[t] = 0.f;

  const int base = head * 128;
  const int kqvn = n * 65536;
  {
    int d = t >> 3, hoff = (t & 7) * 8;              // 8 bf16 per uint4
    int ck = base + d, cq = base + 32 + d;
    U4 kr4, qr4;
    kr4.q = *(const uint4*)&kqv[kqvn + ck * 64 + hoff];
    qr4.q = *(const uint4*)&kqv[kqvn + cq * 64 + hoff];
    float sck = scale[ck], bck = bias[ck], scq = scale[cq], bcq = bias[cq];
#pragma unroll
    for (int j = 0; j < 8; ++j) {
      float kvf = bf2f(kr4.u[j]) * sck + bck;
      float qvf = bf2f(qr4.u[j]) * scq + bcq;
      ks[d * 64 + hoff + j] = kvf;  ksT[(hoff + j) * 32 + d] = f2bf(kvf);
      qs[d * 64 + hoff + j] = qvf;  qsT[(hoff + j) * 32 + d] = f2bf(qvf);
    }
  }
  __syncthreads();

  // qk via WMMA: 16 tiles, 2 per wave
  float s0 = 0.f, s1 = 0.f;
#pragma unroll
  for (int tt = 0; tt < 2; ++tt) {
    int tile = wave * 2 + tt;
    int xt = tile >> 2, yt = tile & 3;
    int xv = xt * 16 + mcol, yv = yt * 16 + mcol;
    v16bf a = ld8x2(&qsT[xv * 32 + kbA], &qsT[xv * 32 + kbA + 8]);
    v16bf bb = ld16(&ksT[yv * 32 + kbB]);
    v8f c = (v8f){};
    c = wmma_bf16(a, bb, c);
#pragma unroll
    for (int r = 0; r < 8; ++r) { float vv = c[r]; s0 += vv; s1 += vv * vv; }
  }
  atomicAdd(&red[0], s0); atomicAdd(&red[1], s1);

  // qr / kr (x-coupled, VALU)
  float s2 = 0.f, s3 = 0.f, s4 = 0.f, s5 = 0.f;
  for (int i = 0; i < 16; ++i) {
    int p = t * 16 + i;
    int xx = p >> 6, yy = p & 63;
    float aq = 0.f, ak = 0.f;
    for (int d = 0; d < 32; ++d) {
      aq += qs[d * 64 + xx] * rel[d * 4096 + xx * 64 + yy];
      ak += ks[d * 64 + yy] * rel[(32 + d) * 4096 + yy * 64 + xx];
    }
    s2 += aq; s3 += aq * aq; s4 += ak; s5 += ak * ak;
  }
  atomicAdd(&red[2], s2); atomicAdd(&red[3], s3);
  atomicAdd(&red[4], s4); atomicAdd(&red[5], s5);
  __syncthreads();
  if (t == 0) {
    atomicAdd(&lsum[head], red[0]);      atomicAdd(&lsq[head], red[1]);
    atomicAdd(&lsum[8 + head], red[2]);  atomicAdd(&lsq[8 + head], red[3]);
    atomicAdd(&lsum[16 + head], red[4]); atomicAdd(&lsq[16 + head], red[5]);
  }
}

// ---------------------------------------------------------------- k_lscale
__global__ void k_lscale(const float* __restrict__ lsum, const float* __restrict__ lsq,
                         const float* __restrict__ lg, float* __restrict__ la) {
  int c = threadIdx.x;
  if (c >= 24) return;
  const float inv = 1.f / (512.f * 4096.f);           // N*H*H
  float m = lsum[c] * inv;
  float var = lsq[c] * inv - m * m;
  la[c] = lg[c] * rsqrtf(var + 1e-5f);                // beta/mean cancel in softmax
}

// ---------------------------------------------------------------- k_attn
// grid (512, 8): block = (n, head). Full attention + output.
__global__ void __launch_bounds__(256) k_attn(
    const unsigned short* __restrict__ kqv, const float* __restrict__ scale,
    const float* __restrict__ bias, const float* __restrict__ rel,
    const float* __restrict__ la, float* __restrict__ out) {
  const int n = blockIdx.x, head = blockIdx.y;
  const int b = n >> 8, w = n & 255;
  const int t = threadIdx.x, lane = t & 31, wave = t >> 5;
  const int mcol = lane & 15, half = lane >> 4;
  const int kbA = (lane < 16) ? 0 : 16;
  const int kbB = (lane < 16) ? 0 : 16;

  __shared__ __attribute__((aligned(32))) float Wl[4096];            // weights [x][y]
  __shared__ __attribute__((aligned(32))) float regA[4096];          // qs+ks -> attn_enc
  __shared__ __attribute__((aligned(32))) unsigned short tbuf[4096]; // qsT+ksT -> Wlb
  __shared__ __attribute__((aligned(32))) unsigned short vsb[4096];  // v [d][y] bf16
  float* qs = regA;                  // [d][h]
  float* ks = regA + 2048;           // [d][h]
  unsigned short* qsT = tbuf;        // [x][d]
  unsigned short* ksT = tbuf + 2048; // [y][d]
  unsigned short* Wlb = tbuf;        // [x][y] bf16 (after softmax)

  const int base = head * 128;
  const int kqvn = n * 65536;
  {
    int d = t >> 3, hoff = (t & 7) * 8;
    int ck = base + d, cq = base + 32 + d;
    U4 kr4, qr4;
    kr4.q = *(const uint4*)&kqv[kqvn + ck * 64 + hoff];
    qr4.q = *(const uint4*)&kqv[kqvn + cq * 64 + hoff];
    float sck = scale[ck], bck = bias[ck], scq = scale[cq], bcq = bias[cq];
#pragma unroll
    for (int j = 0; j < 8; ++j) {
      float kvf = bf2f(kr4.u[j]) * sck + bck;
      float qvf = bf2f(qr4.u[j]) * scq + bcq;
      ks[d * 64 + hoff + j] = kvf;  ksT[(hoff + j) * 32 + d] = f2bf(kvf);
      qs[d * 64 + hoff + j] = qvf;  qsT[(hoff + j) * 32 + d] = f2bf(qvf);
    }
  }
#pragma unroll
  for (int i = 0; i < 2; ++i) {                       // v: 512 chunks of 8 halves
    int c2 = i * 256 + t;
    int d = c2 >> 3, hoff = (c2 & 7) * 8;
    int cv = base + 64 + d;
    U4 r4, o4;
    r4.q = *(const uint4*)&kqv[kqvn + cv * 64 + hoff];
    float sc = scale[cv], bc = bias[cv];
#pragma unroll
    for (int j = 0; j < 8; ++j) o4.u[j] = f2bf(bf2f(r4.u[j]) * sc + bc);
    *(uint4*)&vsb[d * 64 + hoff] = o4.q;
  }
  const float a0 = la[head], a1 = la[8 + head], a2 = la[16 + head];
  __syncthreads();

  // phase 1: qr + kr contributions (exclusive writes)
  for (int i = 0; i < 16; ++i) {
    int p = t * 16 + i;
    int xx = p >> 6, yy = p & 63;
    float aq = 0.f, ak = 0.f;
    for (int d = 0; d < 32; ++d) {
      aq += qs[d * 64 + xx] * rel[d * 4096 + xx * 64 + yy];
      ak += ks[d * 64 + yy] * rel[(32 + d) * 4096 + yy * 64 + xx];
    }
    Wl[p] = a1 * aq + a2 * ak;
  }
  __syncthreads();

  // phase 2: qk via WMMA, add into logits
#pragma unroll
  for (int tt = 0; tt < 2; ++tt) {
    int tile = wave * 2 + tt;
    int xt = tile >> 2, yt = tile & 3;
    int xv = xt * 16 + mcol, yv = yt * 16 + mcol;
    v16bf a = ld8x2(&qsT[xv * 32 + kbA], &qsT[xv * 32 + kbA + 8]);
    v16bf bb = ld16(&ksT[yv * 32 + kbB]);
    v8f c = (v8f){};
    c = wmma_bf16(a, bb, c);
#pragma unroll
    for (int r = 0; r < 8; ++r) {
      int xx = xt * 16 + half * 8 + r;
      Wl[xx * 64 + yv] += a0 * c[r];
    }
  }
  __syncthreads();

  // phase 3: softmax over y, one row per thread
  if (t < 64) {
    float mx = -3.4e38f;
    for (int yy = 0; yy < 64; ++yy) mx = fmaxf(mx, Wl[t * 64 + yy]);
    float s = 0.f;
    for (int yy = 0; yy < 64; ++yy) {
      float e = __expf(Wl[t * 64 + yy] - mx);
      Wl[t * 64 + yy] = e; s += e;
    }
    float inv = 1.f / s;
    for (int yy = 0; yy < 64; ++yy) Wl[t * 64 + yy] *= inv;
  }
  __syncthreads();

  // phase 4: attn_enc (VALU) into regA; mirror weights to bf16 (tbuf reuse)
  for (int i = 0; i < 16; ++i) {
    int p = t * 16 + i;
    int d = p >> 6, xx = p & 63;
    float accv = 0.f;
    for (int yy = 0; yy < 64; ++yy)
      accv += Wl[xx * 64 + yy] * rel[(64 + d) * 4096 + xx * 64 + yy];
    regA[p] = accv;            // [d][x]
    Wlb[p] = f2bf(Wl[p]);      // [x][y] bf16
  }
  __syncthreads();

  // phase 5: attn = W x v^T via WMMA (K=64 -> 2 steps), add enc, store
#pragma unroll
  for (int tt = 0; tt < 2; ++tt) {
    int tile = wave * 2 + tt;
    int dt = tile >> 2, xt = tile & 3;
    int drow = dt * 16 + mcol;
    int xcol = xt * 16 + mcol;
    v8f c = (v8f){};
#pragma unroll
    for (int ks2 = 0; ks2 < 2; ++ks2) {
      v16bf a = ld8x2(&vsb[drow * 64 + ks2 * 32 + kbA],
                      &vsb[drow * 64 + ks2 * 32 + kbA + 8]);
      v16bf bb = ld16(&Wlb[xcol * 64 + ks2 * 32 + kbB]);
      c = wmma_bf16(a, bb, c);
    }
#pragma unroll
    for (int r = 0; r < 8; ++r) {
      int d = dt * 16 + half * 8 + r;
      float val = c[r] + regA[d * 64 + xcol];
      out[((b * 512 + head * 64 + d) * 64 + xcol) * 256 + w] = val;
    }
  }
}

// ---------------------------------------------------------------- launch
extern "C" void kernel_launch(void* const* d_in, const int* in_sizes, int n_in,
                              void* d_out, int out_size, void* d_ws, size_t ws_size,
                              hipStream_t stream) {
  (void)in_sizes; (void)n_in; (void)out_size; (void)ws_size;
  const float* x        = (const float*)d_in[0];
  const float* y        = (const float*)d_in[1];
  const float* k_w      = (const float*)d_in[2];
  const float* q_w      = (const float*)d_in[3];
  const float* v_w      = (const float*)d_in[4];
  const float* k_gamma  = (const float*)d_in[5];
  const float* k_beta   = (const float*)d_in[6];
  const float* q_gamma  = (const float*)d_in[7];
  const float* q_beta   = (const float*)d_in[8];
  const float* kq_gamma = (const float*)d_in[9];
  const float* kq_beta  = (const float*)d_in[10];
  const float* lg_gamma = (const float*)d_in[11];
  // d_in[12] (lg_beta) provably cancels inside softmax; unused.
  const float* rel_enc  = (const float*)d_in[13];
  float* out = (float*)d_out;

  char* wsb = (char*)d_ws;
  unsigned short* kqv = (unsigned short*)wsb;              // 64 MiB (bf16)
  float* rel   = (float*)(wsb + 67108864);                 // 2 MiB
  float* stats = (float*)(wsb + 69206016);                 // 32 KiB region
  float* gsum = stats,        *gsq  = stats + 1024;
  float* lsum = stats + 2048, *lsq  = stats + 2072;
  float* chscale = stats + 2096, *chbias = stats + 3120;
  float* la = stats + 4144;
  unsigned short* wb = (unsigned short*)(wsb + 69238784);  // 1 MiB
  unsigned short* xb = (unsigned short*)(wsb + 70287360);  // 32 MiB
  unsigned short* yb = (unsigned short*)(wsb + 103841792); // 32 MiB

  k_prep  <<<2048, 256, 0, stream>>>(rel_enc, rel, stats);
  k_pack  <<<133120, 256, 0, stream>>>(x, y, k_w, q_w, v_w, wb, xb, yb);
  k_proj  <<<dim3(512, 4), 256, 0, stream>>>(wb, xb, yb, kqv, gsum, gsq);
  k_chstats<<<4, 256, 0, stream>>>(gsum, gsq, k_gamma, k_beta, q_gamma, q_beta,
                                   kq_gamma, kq_beta, chscale, chbias);
  k_lstats<<<dim3(512, 8), 256, 0, stream>>>(kqv, chscale, chbias, rel, lsum, lsq);
  k_lscale<<<1, 32, 0, stream>>>(lsum, lsq, lg_gamma, la);
  k_attn  <<<dim3(512, 8), 256, 0, stream>>>(kqv, chscale, chbias, rel, la, out);
}
